// GenerativeModel_4690104287409
// MI455X (gfx1250) — matrix operand order repaired and checked by reading, not compiled, not run on hardware
//
#include <hip/hip_runtime.h>
#include <hip/hip_bf16.h>
#include <math.h>

// ---------------- problem constants (match reference setup) ----------------
constexpr int H    = 128;           // hidden
constexpr int NB   = 64;            // batch B
constexpr int NPG  = 128;           // nodes per graph
constexpr int NTOT = NB * NPG;      // N = 8192
constexpr int NE   = NTOT * 16;     // E = 131072
constexpr int NREL = 8;             // R
constexpr int NVA  = 20;            // action vocab
constexpr int SEQL = 32;            // L
constexpr int NLAY = 2;

typedef __attribute__((ext_vector_type(2)))  float    v2f;
typedef __attribute__((ext_vector_type(8)))  float    v8f;
typedef __attribute__((ext_vector_type(16))) _Float16 v16h;

#if __has_builtin(__builtin_amdgcn_wmma_f32_16x16x4_f32)
#define USE_F32_WMMA 1
#else
#define USE_F32_WMMA 0
#endif

// ---------------------------------------------------------------------------
// Generic batched WMMA GEMM: C[M,Ncols] = (Cin? Cin:0) + A(MxK) * B(KxNcols)
//  + bias[col], optional relu. One wave handles one 16x16 C tile.
//  TA: A element [m][k] at A[k*lda+m] (else A[m*lda+k])
//  TB: B element [k][n] at B[n*ldb+k] (else B[k*ldb+n])
//  blockIdx.z = batch; sA/sB/sC are per-batch element strides.
// M, Ncols multiples of 16; K multiple of 32.
// ---------------------------------------------------------------------------
template <int TA, int TB>
__global__ void k_gemm_wmma(float* __restrict__ Cout, const float* __restrict__ Cin,
                            const float* __restrict__ A, const float* __restrict__ Bm,
                            const float* __restrict__ bias,
                            int M, int Ncols, int K, int lda, int ldb, int ldc,
                            long sA, long sB, long sC, int relu) {
  int wave   = (blockIdx.x * blockDim.x + threadIdx.x) >> 5;
  int tilesN = Ncols >> 4;
  int nTiles = (M >> 4) * tilesN;
  if (wave >= nTiles) return;                    // wave-uniform exit (EXEC stays all-1 in WMMA)
  int lane = threadIdx.x & 31;
  int half = lane >> 4;                          // 0: lanes 0-15, 1: lanes 16-31
  int idx  = lane & 15;
  int m0 = (wave / tilesN) << 4;
  int n0 = (wave % tilesN) << 4;
  const float* Ab = A  + (long)blockIdx.z * sA;
  const float* Bb = Bm + (long)blockIdx.z * sB;
  float*       Co = Cout + (long)blockIdx.z * sC;

  // C/D layout (ISA 7.12.2): VGPR v, lanes 0-15 -> M=v, lanes 16-31 -> M=v+8
  v8f acc;
  if (Cin) {
    const float* Ci = Cin + (long)blockIdx.z * sC;
#pragma unroll
    for (int v = 0; v < 8; v++) acc[v] = Ci[(long)(m0 + v + 8 * half) * ldc + n0 + idx];
  } else {
#pragma unroll
    for (int v = 0; v < 8; v++) acc[v] = 0.0f;
  }

#if USE_F32_WMMA
  // fp32 tensor path: V_WMMA_F32_16X16X4_F32, K step 4.
  // A 16x4 f32 layout: lane m=idx; VGPR j holds K = 2*half + j.
#pragma unroll 4
  for (int k0 = 0; k0 < K; k0 += 4) {
    int ka = k0 + 2 * half;
    v2f av, bv;
    av.x = TA ? Ab[(long)(ka + 0) * lda + (m0 + idx)] : Ab[(long)(m0 + idx) * lda + (ka + 0)];
    av.y = TA ? Ab[(long)(ka + 1) * lda + (m0 + idx)] : Ab[(long)(m0 + idx) * lda + (ka + 1)];
    bv.x = TB ? Bb[(long)(n0 + idx) * ldb + (ka + 0)] : Bb[(long)(ka + 0) * ldb + (n0 + idx)];
    bv.y = TB ? Bb[(long)(n0 + idx) * ldb + (ka + 1)] : Bb[(long)(ka + 1) * ldb + (n0 + idx)];
    acc = __builtin_amdgcn_wmma_f32_16x16x4_f32(false, av, false, bv, (short)0, acc,
                                                false, false);
  }
#else
  // fallback: V_WMMA_F32_16X16X32_F16 with on-the-fly f32->f16 converts.
  for (int k0 = 0; k0 < K; k0 += 32) {
    v16h av, bv;
#pragma unroll
    for (int e = 0; e < 16; e++) {
      int v = e >> 1, j = e & 1;
      // A 16x32 f16 layout: VGPR0-3: K=0..7 (lanes<16) / 8..15 ; VGPR4-7: +16
      int ka = k0 + ((v < 4) ? (2 * v + j + 8 * half) : (16 + 2 * (v - 4) + j + 8 * half));
      av[e] = (_Float16)(TA ? Ab[(long)ka * lda + (m0 + idx)]
                            : Ab[(long)(m0 + idx) * lda + ka]);
      // B 32x16 f16 layout: same K mapping as A
      int kb = ka;
      bv[e] = (_Float16)(TB ? Bb[(long)(n0 + idx) * ldb + kb]
                            : Bb[(long)kb * ldb + (n0 + idx)]);
    }
    acc = __builtin_amdgcn_wmma_f32_16x16x32_f16(false, av, false, bv, (short)0, acc,
                                                 false, false);
  }
#endif

  float bval = bias ? bias[n0 + idx] : 0.0f;
#pragma unroll
  for (int v = 0; v < 8; v++) {
    float x = acc[v] + bval;
    if (relu) x = fmaxf(x, 0.0f);
    Co[(long)(m0 + v + 8 * half) * ldc + n0 + idx] = x;
  }
}

// ------------------------------- helper kernels ----------------------------
__global__ void k_zero(float* p, int n) {
  int i = blockIdx.x * blockDim.x + threadIdx.x;
  if (i < n) p[i] = 0.0f;
}

__global__ void k_copy(const float* a, float* b, int n) {
  int i = blockIdx.x * blockDim.x + threadIdx.x;
  if (i < n) b[i] = a[i];
}

__global__ void k_embed(const int* __restrict__ types, const float* __restrict__ emb,
                        float* __restrict__ x) {
  int i = blockIdx.x * blockDim.x + threadIdx.x;
  if (i < NTOT * H) {
    int node = i >> 7, h = i & 127;
    x[i] = emb[types[node] * H + h];
  }
}

__global__ void k_count(const int* __restrict__ etype, const int* __restrict__ dst,
                        float* __restrict__ cnt) {
  int e = blockIdx.x * blockDim.x + threadIdx.x;
  if (e < NE) atomicAdd(&cnt[etype[e] * NTOT + dst[e]], 1.0f);
}

// scatter x[src] rows into agg[dst] for edges of relation r; 1 wave per edge,
// each lane handles 4 consecutive floats of the 128-float row.
__global__ void k_scatter(const float* __restrict__ x, const int* __restrict__ src,
                          const int* __restrict__ dst, const int* __restrict__ etype,
                          int r, float* __restrict__ agg) {
  int t = blockIdx.x * blockDim.x + threadIdx.x;
  int e = t >> 5;
  if (e >= NE) return;
  if (etype[e] != r) return;
  int h0 = (t & 31) * 4;
  int s = src[e] * H + h0;
  int d = dst[e] * H + h0;
  atomicAdd(&agg[d + 0], x[s + 0]);
  atomicAdd(&agg[d + 1], x[s + 1]);
  atomicAdd(&agg[d + 2], x[s + 2]);
  atomicAdd(&agg[d + 3], x[s + 3]);
}

__global__ void k_normalize(float* __restrict__ agg, const float* __restrict__ cntR) {
  int i = blockIdx.x * blockDim.x + threadIdx.x;
  if (i < NTOT * H) agg[i] /= fmaxf(cntR[i >> 7], 1.0f);
}

__global__ void k_meanpool(const float* __restrict__ x, float* __restrict__ hG) {
  int i = blockIdx.x * blockDim.x + threadIdx.x;
  if (i < NB * H) {
    int b = i >> 7, h = i & 127;
    const float* p = x + (long)(b * NPG) * H + h;
    float s = 0.0f;
    for (int j = 0; j < NPG; j++) s += p[j * H];
    hG[i] = s * (1.0f / (float)NPG);
  }
}

// out[b,o] = act(sum_k in[b,k]*W[o,k] + bias[o]); mode 1 => sigmoid
__global__ void k_head(const float* __restrict__ in, const float* __restrict__ W,
                       const float* __restrict__ bias, float* __restrict__ outp,
                       int O, int K, int mode) {
  int i = blockIdx.x * blockDim.x + threadIdx.x;
  if (i >= NB * O) return;
  int b = i / O, o = i % O;
  float acc = bias[o];
  const float* row = in + b * K;
  const float* wro = W + o * K;
  for (int k = 0; k < K; k++) acc += row[k] * wro[k];
  if (mode == 1) acc = 1.0f / (1.0f + expf(-acc));
  outp[i] = acc;
}

// gx[b,t,:] = t==0 ? emb_actions[act[b]] : oseq[b,t-1,:]
__global__ void k_gruinput(const float* __restrict__ oseq, const float* __restrict__ embA,
                           const int* __restrict__ act, float* __restrict__ gx) {
  int i = blockIdx.x * blockDim.x + threadIdx.x;
  if (i >= NB * SEQL * H) return;
  int h = i & 127;
  int t = (i >> 7) & 31;
  int b = i >> 12;
  gx[i] = (t == 0) ? embA[act[b] * H + h] : oseq[(((b * SEQL) + (t - 1)) << 7) + h];
}

__global__ void k_gru_gate(const float* __restrict__ gi, const float* __restrict__ gh,
                           float* __restrict__ h, float* __restrict__ gout,
                           const int* __restrict__ len, int t) {
  int i = blockIdx.x * blockDim.x + threadIdx.x;
  if (i >= NB * H) return;
  int b = i >> 7, j = i & 127;
  const float* gib = gi + b * 3 * H;
  const float* ghb = gh + b * 3 * H;
  float r  = 1.0f / (1.0f + expf(-(gib[j] + ghb[j])));
  float z  = 1.0f / (1.0f + expf(-(gib[H + j] + ghb[H + j])));
  float nn = tanhf(gib[2 * H + j] + r * ghb[2 * H + j]);
  float hv = h[i];
  float hn = (1.0f - z) * nn + z * hv;
  bool valid = t < len[b];
  h[i] = valid ? hn : hv;
  gout[(((b * SEQL) + t) << 7) + j] = valid ? hn : 0.0f;
}

// logits[n,l] = sum_h relu(a1[b(n),l,h] + c2[n,h]) * wf[h] + bf
// one wave per node; lane = l (L == 32 == wave size)
__global__ void k_logits(const float* __restrict__ a1, const float* __restrict__ c2,
                         const float* __restrict__ wf, const float* __restrict__ bf,
                         float* __restrict__ outp) {
  int w = (blockIdx.x * blockDim.x + threadIdx.x) >> 5;
  if (w >= NTOT) return;
  int lane = threadIdx.x & 31;
  int b = w >> 7;
  const float* a = a1 + (long)((b * SEQL + lane)) * H;
  const float* c = c2 + (long)w * H;
  float acc = 0.0f;
#pragma unroll 4
  for (int h = 0; h < H; h++) acc += fmaxf(a[h] + c[h], 0.0f) * wf[h];
  outp[w * SEQL + lane] = acc + bf[0];
}

// in-place segment softmax over contiguous groups of NPG nodes, per column l.
// one wave per (b,l); each lane covers 4 nodes.
__global__ void k_softmax(float* __restrict__ p) {
  int w = (blockIdx.x * blockDim.x + threadIdx.x) >> 5;
  if (w >= NB * SEQL) return;
  int lane = threadIdx.x & 31;
  int b = w / SEQL, l = w % SEQL;
  float v[4];
  int   ix[4];
#pragma unroll
  for (int j = 0; j < 4; j++) {
    int node = b * NPG + lane + 32 * j;
    ix[j] = node * SEQL + l;
    v[j] = p[ix[j]];
  }
  float m = fmaxf(fmaxf(v[0], v[1]), fmaxf(v[2], v[3]));
  for (int o = 16; o >= 1; o >>= 1) m = fmaxf(m, __shfl_xor(m, o, 32));
  float s = 0.0f;
#pragma unroll
  for (int j = 0; j < 4; j++) { v[j] = expf(v[j] - m); s += v[j]; }
  for (int o = 16; o >= 1; o >>= 1) s += __shfl_xor(s, o, 32);
  float inv = 1.0f / s;
#pragma unroll
  for (int j = 0; j < 4; j++) p[ix[j]] = v[j] * inv;
}

// ------------------------------- host helpers ------------------------------
static inline int b256(int n) { return (n + 255) / 256; }

static void gemm(float* Cout, const float* Cin, const float* A, const float* Bm,
                 const float* bias, int M, int Ncols, int K,
                 int lda, int ldb, int ldc, bool tA, bool tB,
                 long sA, long sB, long sC, int batch, bool relu, hipStream_t s) {
  int tiles = (M / 16) * (Ncols / 16);
  dim3 grid((tiles + 7) / 8, 1, batch);
  if (!tA && !tB)
    k_gemm_wmma<0, 0><<<grid, 256, 0, s>>>(Cout, Cin, A, Bm, bias, M, Ncols, K,
                                           lda, ldb, ldc, sA, sB, sC, (int)relu);
  else if (!tA && tB)
    k_gemm_wmma<0, 1><<<grid, 256, 0, s>>>(Cout, Cin, A, Bm, bias, M, Ncols, K,
                                           lda, ldb, ldc, sA, sB, sC, (int)relu);
  else if (tA && !tB)
    k_gemm_wmma<1, 0><<<grid, 256, 0, s>>>(Cout, Cin, A, Bm, bias, M, Ncols, K,
                                           lda, ldb, ldc, sA, sB, sC, (int)relu);
  else
    k_gemm_wmma<1, 1><<<grid, 256, 0, s>>>(Cout, Cin, A, Bm, bias, M, Ncols, K,
                                           lda, ldb, ldc, sA, sB, sC, (int)relu);
}

// ------------------------------- entry point -------------------------------
extern "C" void kernel_launch(void* const* d_in, const int* in_sizes, int n_in,
                              void* d_out, int out_size, void* d_ws, size_t ws_size,
                              hipStream_t stream) {
  (void)in_sizes; (void)n_in; (void)out_size; (void)ws_size;

  const int*   nodeTypes = (const int*)d_in[0];
  const int*   eidx      = (const int*)d_in[1];
  const int*   esrc      = eidx;
  const int*   edst      = eidx + NE;
  const int*   etype     = (const int*)d_in[2];
  // d_in[3]=bs (groups are contiguous by construction), d_in[4]=nodes_bs (==NPG)
  const float* seq       = (const float*)d_in[5];
  const int*   lens      = (const int*)d_in[6];
  const int*   act       = (const int*)d_in[7];
  const float* embN      = (const float*)d_in[8];
  const float* embA      = (const float*)d_in[9];
  const float* Wrel      = (const float*)d_in[10];   // [2,8,H,H]
  const float* Wroot     = (const float*)d_in[11];   // [2,H,H]
  const float* rb        = (const float*)d_in[12];   // [2,H]
  const float* Wih       = (const float*)d_in[13];   // [3H,H]
  const float* Whh       = (const float*)d_in[14];   // [3H,H]
  const float* bih       = (const float*)d_in[15];
  const float* bhh       = (const float*)d_in[16];
  const float* linA_W    = (const float*)d_in[17];
  const float* linA_b    = (const float*)d_in[18];
  const float* linAf_W   = (const float*)d_in[19];
  const float* linAf_b   = (const float*)d_in[20];
  const float* finL_W    = (const float*)d_in[21];
  const float* finL_b    = (const float*)d_in[22];
  const float* finF_W    = (const float*)d_in[23];
  const float* finF_b    = (const float*)d_in[24];
  const float* linN_W    = (const float*)d_in[25];   // [H,2H]
  const float* linN_b    = (const float*)d_in[26];
  const float* linNf_W   = (const float*)d_in[27];   // [1,H]
  const float* linNf_b   = (const float*)d_in[28];

  float* outAction = (float*)d_out;                    // [B,VA]
  float* outNodes  = (float*)d_out + NB * NVA;         // [N,L]
  float* outFinal  = (float*)d_out + NB * NVA + NTOT * SEQL;  // [B]

  // workspace carve-up (floats)
  float* ws  = (float*)d_ws;
  float* x0  = ws;  ws += NTOT * H;            // node features (ping)
  float* x1  = ws;  ws += NTOT * H;            // node features (pong)
  float* ob  = ws;  ws += NTOT * H;            // RGCN accumulation buffer
  float* agg = ws;  ws += NTOT * H;            // per-relation scatter buffer
  float* cnt = ws;  ws += NREL * NTOT;         // per-(r,node) edge counts
  float* hG  = ws;  ws += NB * H;
  float* tmp = ws;  ws += NB * H;
  float* osq = ws;  ws += NB * SEQL * H;       // per-batch S^T @ X
  float* gx  = ws;  ws += NB * SEQL * H;       // GRU input sequence
  float* hst = ws;  ws += NB * H;              // GRU hidden state
  float* gi  = ws;  ws += NB * 3 * H;
  float* gh  = ws;  ws += NB * 3 * H;
  float* go  = ws;  ws += NB * SEQL * H;       // GRU outputs (padded with 0)
  float* a1  = ws;  ws += NB * SEQL * H;       // gru@W1^T + linN_b
  float* c2  = ws;  ws += NTOT * H;            // emb@W2^T

  // 1) node-type embedding
  k_embed<<<b256(NTOT * H), 256, 0, stream>>>(nodeTypes, embN, x0);

  // 2) per-(relation,node) in-degree counts (same for both layers)
  k_zero<<<b256(NREL * NTOT), 256, 0, stream>>>(cnt, NREL * NTOT);
  k_count<<<b256(NE), 256, 0, stream>>>(etype, edst, cnt);

  // 3) two RGCN layers: out = relu(x@Wroot + b + sum_r mean_r(x[src])@Wrel[r])
  for (int l = 0; l < NLAY; l++) {
    const float* xin = (l == 0) ? x0 : x1;
    float*       xout = (l == 0) ? x1 : x0;
    gemm(ob, nullptr, xin, Wroot + l * H * H, rb + l * H,
         NTOT, H, H, H, H, H, false, false, 0, 0, 0, 1, false, stream);
    for (int r = 0; r < NREL; r++) {
      k_zero<<<b256(NTOT * H), 256, 0, stream>>>(agg, NTOT * H);
      k_scatter<<<b256(NE * 32), 256, 0, stream>>>(xin, esrc, edst, etype, r, agg);
      k_normalize<<<b256(NTOT * H), 256, 0, stream>>>(agg, cnt + r * NTOT);
      bool last = (r == NREL - 1);
      gemm(last ? xout : ob, ob, agg, Wrel + (l * NREL + r) * H * H, nullptr,
           NTOT, H, H, H, H, H, false, false, 0, 0, 0, 1, last, stream);
    }
  }
  const float* nodeEmb = x0;   // after layer 1, result ping-pongs back to x0

  // 4) graph mean pooling
  k_meanpool<<<b256(NB * H), 256, 0, stream>>>(nodeEmb, hG);

  // 5) action head: relu(hG@linA_W^T+b) @ linAf_W^T + b
  gemm(tmp, nullptr, hG, linA_W, linA_b, NB, H, H, H, H, H,
       false, true, 0, 0, 0, 1, true, stream);
  k_head<<<b256(NB * NVA), 256, 0, stream>>>(tmp, linAf_W, linAf_b, outAction, NVA, H, 0);

  // 6) final head: sigmoid(relu(hG@finL_W^T+b) @ finF_W^T + b)
  gemm(tmp, nullptr, hG, finL_W, finL_b, NB, H, H, H, H, H,
       false, true, 0, 0, 0, 1, true, stream);
  k_head<<<b256(NB), 256, 0, stream>>>(tmp, finF_W, finF_b, outFinal, 1, H, 1);

  // 7) osq[b] = seq_b^T (L x NPG) @ emb_b (NPG x H)  -- batched WMMA
  gemm(osq, nullptr, seq, nodeEmb, nullptr, SEQL, H, NPG,
       SEQL, H, H, true, false, (long)NPG * SEQL, (long)NPG * H, (long)SEQL * H,
       NB, false, stream);

  // 8) GRU over L steps (packed semantics: state frozen / output zero past len)
  k_gruinput<<<b256(NB * SEQL * H), 256, 0, stream>>>(osq, embA, act, gx);
  k_copy<<<b256(NB * H), 256, 0, stream>>>(hG, hst, NB * H);
  for (int t = 0; t < SEQL; t++) {
    gemm(gi, nullptr, gx + t * H, Wih, bih, NB, 3 * H, H,
         SEQL * H, H, 3 * H, false, true, 0, 0, 0, 1, false, stream);
    gemm(gh, nullptr, hst, Whh, bhh, NB, 3 * H, H,
         H, H, 3 * H, false, true, 0, 0, 0, 1, false, stream);
    k_gru_gate<<<b256(NB * H), 256, 0, stream>>>(gi, gh, hst, go, lens, t);
  }

  // 9) rank-split of concat([out_gru, nodeEmb]) @ linN_W^T:
  //    a1 = gru_out @ W1^T + linN_b ;  c2 = nodeEmb @ W2^T
  gemm(a1, nullptr, go, linN_W, linN_b, NB * SEQL, H, H,
       H, 2 * H, H, false, true, 0, 0, 0, 1, false, stream);
  gemm(c2, nullptr, nodeEmb, linN_W + H, nullptr, NTOT, H, H,
       H, 2 * H, H, false, true, 0, 0, 0, 1, false, stream);

  // 10) fused logits + segment softmax (in-place in d_out)
  k_logits<<<NTOT / 8, 256, 0, stream>>>(a1, c2, linNf_W, linNf_b, outNodes);
  k_softmax<<<(NB * SEQL) / 8, 256, 0, stream>>>(outNodes);
}
